// GAT_59150289600576
// MI455X (gfx1250) — matrix operand order repaired and checked by reading, not compiled, not run on hardware
//
#include <hip/hip_runtime.h>

#define D_IN   256
#define D_OUT  64
#define NODES  4
#define LRELU_ALPHA 0.2f
#define NEG_INF -9.0e15f

typedef __attribute__((ext_vector_type(16))) _Float16 v16h;
typedef __attribute__((ext_vector_type(8)))  _Float16 h8;
typedef __attribute__((ext_vector_type(8)))  float    v8f;

#define WT_STRIDE 264   // 256 + 8 halfs pad -> 16 lanes land on distinct bank groups
#define W2_STRIDE 72    // 64 + 8 halfs pad
#define T_STRIDE  72    // per-wave 16x64 tile, padded

// ---------------- prep: W^T (f16) and folded W2eff^T (f16) into workspace ---------
__global__ void gat_prep(const float* __restrict__ W,    // (256,64)
                         const float* __restrict__ W2,   // (512,64)
                         _Float16* __restrict__ Wt,      // (64,256)  Wt[n][k] = W[k][n]
                         _Float16* __restrict__ W2t)     // (64,64)   W2t[n][k] = sum_h W2[64h+k][n]
{
    int tid = blockIdx.x * blockDim.x + threadIdx.x;
    int stride = gridDim.x * blockDim.x;
    for (int i = tid; i < D_OUT * D_IN; i += stride) {
        int n = i >> 8, k = i & 255;
        Wt[i] = (_Float16)W[k * D_OUT + n];
    }
    for (int i = tid; i < D_OUT * D_OUT; i += stride) {
        int n = i >> 6, k = i & 63;
        float s = 0.f;
        #pragma unroll
        for (int hh = 0; hh < 8; ++hh) s += W2[(hh * 64 + k) * D_OUT + n];
        W2t[i] = (_Float16)s;
    }
}

// ---------------- fused GAT: both layers + attention + fc, one pass over x -------
__global__ __launch_bounds__(256) void gat_fused(
    const float* __restrict__ x,    // (B,4,256)
    const float* __restrict__ adj,  // (B,4,4)
    const float* __restrict__ a,    // (128)
    const float* __restrict__ a2,   // (128)
    const float* __restrict__ fcw,  // (64)
    const float* __restrict__ fcb,  // (1)
    const _Float16* __restrict__ Wt,
    const _Float16* __restrict__ W2t,
    float* __restrict__ out, int Btotal)
{
    __shared__ _Float16 sWt [64 * WT_STRIDE];     // 33792 B
    __shared__ _Float16 sW2t[64 * W2_STRIDE];     //  9216 B
    __shared__ _Float16 sT[8][16 * T_STRIDE];     // 18432 B (per-wave staging)

    const int tid = threadIdx.x;

    // preload weights into padded LDS (uint4 = 8 halfs per copy)
    for (int i = tid; i < 64 * 32; i += 256) {
        int r = i >> 5, q = i & 31;
        ((uint4*)(sWt + r * WT_STRIDE))[q] = ((const uint4*)(Wt + r * 256))[q];
    }
    for (int i = tid; i < 64 * 8; i += 256) {
        int r = i >> 3, q = i & 7;
        ((uint4*)(sW2t + r * W2_STRIDE))[q] = ((const uint4*)(W2t + r * 64))[q];
    }
    __syncthreads();

    const int wave = tid >> 5;
    const int lane = tid & 31;
    const int g0 = blockIdx.x * 32 + wave * 4;   // first of this wave's 4 graphs
    if (g0 >= Btotal) return;                    // uniform per wave

    // ---- WMMA fragment coordinates (16-bit A layout / B layout, wave32) ----
    const int arow = lane & 15;                  // A: row M
    const int akb  = (lane < 16) ? 0 : 8;        // A: K sub-base
    const int bcol = lane & 15;                  // B: column N
    const int bkb  = (lane < 16) ? 0 : 8;        // B: K sub-base
    _Float16* tw = sT[wave];

    // =================== layer 1: Wh = X(16x256) @ W(256x64) ===================
    const float* xrow = x + (size_t)(g0 * NODES + arow) * D_IN;
    v8f acc0 = {}, acc1 = {}, acc2 = {}, acc3 = {};
    #pragma unroll
    for (int ks = 0; ks < 8; ++ks) {
        const int ka = ks * 32 + akb;
        float4 f0 = ((const float4*)(xrow + ka))[0];
        float4 f1 = ((const float4*)(xrow + ka))[1];
        float4 f2 = ((const float4*)(xrow + ka + 16))[0];
        float4 f3 = ((const float4*)(xrow + ka + 16))[1];
        float xv[16] = { f0.x, f0.y, f0.z, f0.w, f1.x, f1.y, f1.z, f1.w,
                         f2.x, f2.y, f2.z, f2.w, f3.x, f3.y, f3.z, f3.w };
        v16h af;
        #pragma unroll
        for (int i = 0; i < 16; ++i) af[i] = (_Float16)xv[i];
        const int kb = ks * 32 + bkb;
        #define L1_TILE(N, ACC) do {                                              \
            const _Float16* wp = sWt + ((N) * 16 + bcol) * WT_STRIDE + kb;        \
            h8 blo = *(const h8*)wp;                                              \
            h8 bhi = *(const h8*)(wp + 16);                                       \
            v16h bf = __builtin_shufflevector(blo, bhi, 0,1,2,3,4,5,6,7,          \
                                              8,9,10,11,12,13,14,15);             \
            ACC = __builtin_amdgcn_wmma_f32_16x16x32_f16(false, af, false, bf,    \
                                              (short)0, ACC, false, false);       \
        } while (0)
        L1_TILE(0, acc0); L1_TILE(1, acc1); L1_TILE(2, acc2); L1_TILE(3, acc3);
        #undef L1_TILE
    }

    // store Wh (f16) to per-wave LDS tile; C-layout: lanes 0-15 -> M=j, 16-31 -> M=j+8
    const int crow = (lane < 16) ? 0 : 8;
    const int ccol = lane & 15;
    #pragma unroll
    for (int j = 0; j < 8; ++j) {
        tw[(crow + j) * T_STRIDE +  0 + ccol] = (_Float16)acc0[j];
        tw[(crow + j) * T_STRIDE + 16 + ccol] = (_Float16)acc1[j];
        tw[(crow + j) * T_STRIDE + 32 + ccol] = (_Float16)acc2[j];
        tw[(crow + j) * T_STRIDE + 48 + ccol] = (_Float16)acc3[j];
    }

    // =================== layer-1 attention (8 lanes / graph) ===================
    const int lg = lane >> 3;          // graph within wave
    const int lc = (lane & 7) * 8;     // this lane's 8-channel slice
    const int b  = g0 + lg;

    float whr[4][8];
    #pragma unroll
    for (int n = 0; n < 4; ++n) {
        h8 hv = *(const h8*)(tw + (lg * 4 + n) * T_STRIDE + lc);
        #pragma unroll
        for (int c = 0; c < 8; ++c) whr[n][c] = (float)hv[c];
    }
    float asrc[8], adst[8];
    #pragma unroll
    for (int c = 0; c < 8; ++c) { asrc[c] = a[lc + c]; adst[c] = a[64 + lc + c]; }

    float fs[4], fd[4];
    #pragma unroll
    for (int n = 0; n < 4; ++n) {
        float s = 0.f, d = 0.f;
        #pragma unroll
        for (int c = 0; c < 8; ++c) { s += whr[n][c] * asrc[c]; d += whr[n][c] * adst[c]; }
        s += __shfl_xor(s, 1); s += __shfl_xor(s, 2); s += __shfl_xor(s, 4);
        d += __shfl_xor(d, 1); d += __shfl_xor(d, 2); d += __shfl_xor(d, 4);
        fs[n] = s; fd[n] = d;
    }

    float adjv[16];
    {
        const float4* ap = (const float4*)(adj + (size_t)b * 16);
        float4 r0 = ap[0], r1 = ap[1], r2 = ap[2], r3 = ap[3];
        adjv[0]=r0.x; adjv[1]=r0.y; adjv[2]=r0.z; adjv[3]=r0.w;
        adjv[4]=r1.x; adjv[5]=r1.y; adjv[6]=r1.z; adjv[7]=r1.w;
        adjv[8]=r2.x; adjv[9]=r2.y; adjv[10]=r2.z; adjv[11]=r2.w;
        adjv[12]=r3.x; adjv[13]=r3.y; adjv[14]=r3.z; adjv[15]=r3.w;
    }

    float att[4][4];
    #pragma unroll
    for (int i = 0; i < 4; ++i) {
        float e[4];
        #pragma unroll
        for (int j = 0; j < 4; ++j) {
            float t = fs[i] + fd[j];
            t = (t > 0.f) ? t : LRELU_ALPHA * t;
            e[j] = (adjv[i * 4 + j] > 0.f) ? t : NEG_INF;
        }
        float m = fmaxf(fmaxf(e[0], e[1]), fmaxf(e[2], e[3]));
        float p0 = __expf(e[0] - m), p1 = __expf(e[1] - m);
        float p2 = __expf(e[2] - m), p3 = __expf(e[3] - m);
        float inv = 1.f / (p0 + p1 + p2 + p3);
        att[i][0] = p0 * inv; att[i][1] = p1 * inv;
        att[i][2] = p2 * inv; att[i][3] = p3 * inv;
    }

    // h1 = elu(att @ Wh); overwrite own slice of the wave tile with f16 h1
    #pragma unroll
    for (int n = 0; n < 4; ++n) {
        h8 hv;
        #pragma unroll
        for (int c = 0; c < 8; ++c) {
            float v = att[n][0] * whr[0][c] + att[n][1] * whr[1][c]
                    + att[n][2] * whr[2][c] + att[n][3] * whr[3][c];
            v = (v > 0.f) ? v : (__expf(v) - 1.f);
            hv[c] = (_Float16)v;
        }
        *(h8*)(tw + (lg * 4 + n) * T_STRIDE + lc) = hv;
    }

    // =================== layer 2: Wh2 = h1(16x64) @ W2eff(64x64) ===============
    v8f d0 = {}, d1 = {}, d2 = {}, d3 = {};
    const _Float16* trow = tw + arow * T_STRIDE;
    #pragma unroll
    for (int ks = 0; ks < 2; ++ks) {
        const int ka = ks * 32 + akb;
        h8 alo = *(const h8*)(trow + ka);
        h8 ahi = *(const h8*)(trow + ka + 16);
        v16h af = __builtin_shufflevector(alo, ahi, 0,1,2,3,4,5,6,7,
                                          8,9,10,11,12,13,14,15);
        const int kb = ks * 32 + bkb;
        #define L2_TILE(N, ACC) do {                                              \
            const _Float16* wp = sW2t + ((N) * 16 + bcol) * W2_STRIDE + kb;       \
            h8 blo = *(const h8*)wp;                                              \
            h8 bhi = *(const h8*)(wp + 16);                                       \
            v16h bf = __builtin_shufflevector(blo, bhi, 0,1,2,3,4,5,6,7,          \
                                              8,9,10,11,12,13,14,15);             \
            ACC = __builtin_amdgcn_wmma_f32_16x16x32_f16(false, af, false, bf,    \
                                              (short)0, ACC, false, false);       \
        } while (0)
        L2_TILE(0, d0); L2_TILE(1, d1); L2_TILE(2, d2); L2_TILE(3, d3);
        #undef L2_TILE
    }
    #pragma unroll
    for (int j = 0; j < 8; ++j) {
        tw[(crow + j) * T_STRIDE +  0 + ccol] = (_Float16)d0[j];
        tw[(crow + j) * T_STRIDE + 16 + ccol] = (_Float16)d1[j];
        tw[(crow + j) * T_STRIDE + 32 + ccol] = (_Float16)d2[j];
        tw[(crow + j) * T_STRIDE + 48 + ccol] = (_Float16)d3[j];
    }

    // =========== layer-2 attention (row 0 only) + fc head ===========
    float w2r[4][8];
    #pragma unroll
    for (int n = 0; n < 4; ++n) {
        h8 hv = *(const h8*)(tw + (lg * 4 + n) * T_STRIDE + lc);
        #pragma unroll
        for (int c = 0; c < 8; ++c) w2r[n][c] = (float)hv[c];
    }
    float a2s[8], a2d[8], fcv[8];
    #pragma unroll
    for (int c = 0; c < 8; ++c) {
        a2s[c] = a2[lc + c]; a2d[c] = a2[64 + lc + c]; fcv[c] = fcw[lc + c];
    }
    float fs0 = 0.f;
    #pragma unroll
    for (int c = 0; c < 8; ++c) fs0 += w2r[0][c] * a2s[c];
    fs0 += __shfl_xor(fs0, 1); fs0 += __shfl_xor(fs0, 2); fs0 += __shfl_xor(fs0, 4);

    float fd2[4], vv[4];
    #pragma unroll
    for (int n = 0; n < 4; ++n) {
        float d = 0.f, v = 0.f;
        #pragma unroll
        for (int c = 0; c < 8; ++c) { d += w2r[n][c] * a2d[c]; v += w2r[n][c] * fcv[c]; }
        d += __shfl_xor(d, 1); d += __shfl_xor(d, 2); d += __shfl_xor(d, 4);
        v += __shfl_xor(v, 1); v += __shfl_xor(v, 2); v += __shfl_xor(v, 4);
        fd2[n] = d; vv[n] = v;
    }
    float e2[4];
    #pragma unroll
    for (int m = 0; m < 4; ++m) {
        float t = fs0 + fd2[m];
        t = (t > 0.f) ? t : LRELU_ALPHA * t;
        e2[m] = (adjv[m] > 0.f) ? t : NEG_INF;   // adj row 0
    }
    float mx = fmaxf(fmaxf(e2[0], e2[1]), fmaxf(e2[2], e2[3]));
    float q0 = __expf(e2[0] - mx), q1 = __expf(e2[1] - mx);
    float q2 = __expf(e2[2] - mx), q3 = __expf(e2[3] - mx);
    float inv = 1.f / (q0 + q1 + q2 + q3);
    float res = (q0 * vv[0] + q1 * vv[1] + q2 * vv[2] + q3 * vv[3]) * inv + fcb[0];

    if ((lane & 7) == 0) out[b] = res;
}

extern "C" void kernel_launch(void* const* d_in, const int* in_sizes, int n_in,
                              void* d_out, int out_size, void* d_ws, size_t ws_size,
                              hipStream_t stream) {
    const float* x   = (const float*)d_in[0];
    const float* adj = (const float*)d_in[1];
    const float* W   = (const float*)d_in[2];
    const float* a   = (const float*)d_in[3];
    const float* W2  = (const float*)d_in[4];
    const float* a2  = (const float*)d_in[5];
    const float* fcw = (const float*)d_in[6];
    const float* fcb = (const float*)d_in[7];
    float* out = (float*)d_out;

    const int Btotal = in_sizes[0] / (NODES * D_IN);

    _Float16* Wt  = (_Float16*)d_ws;          // 64*256 f16 = 32 KB
    _Float16* W2t = Wt + 64 * 256;            // 64*64  f16 =  8 KB

    gat_prep<<<16, 256, 0, stream>>>(W, W2, Wt, W2t);
    const int blocks = (Btotal + 31) / 32;    // 32 graphs per block (8 waves x 4)
    gat_fused<<<blocks, 256, 0, stream>>>(x, adj, a, a2, fcw, fcb, Wt, W2t, out, Btotal);
}
